// MolecularMultiHeadedAttention_48885317763248
// MI455X (gfx1250) — matrix operand order repaired and checked by reading, not compiled, not run on hardware
//
#include <hip/hip_runtime.h>
#include <hip/hip_bf16.h>

// MolecularMultiHeadedAttention for gfx1250 (MI455X), bf16 WMMA pipeline.
// B=16, N=512, D=1024, H=16, DK=64.
//
//   x = 0.3*softmax(QK^T/sqrt(dk), mask)@V  +  (0.3*p_dist + 0.4*p_adj)@V
//   out = x @ Wo^T + bo
//
// bf16 WMMA (16x16x32, f32 accumulate); V stored pre-transposed [B,D,N] so all
// GEMM/attention tile fills are contiguous async global->LDS copies (ASYNCcnt),
// double-buffered; flash-style online softmax in the attention kernel.

#define Bb 16
#define Nn 512
#define Dd 1024
#define Hh 16
#define DKk 64

typedef __attribute__((ext_vector_type(16))) __bf16 v16bf;
typedef __attribute__((ext_vector_type(8)))  float  v8f;

#define WMMA_BF16(a, b, c) \
  __builtin_amdgcn_wmma_f32_16x16x32_bf16(false, (a), false, (b), (short)0, (c), false, false)

// ---------------------------------------------------------------- helpers ---

static __device__ __forceinline__ unsigned short f2bf(float f) {
  union { float f; unsigned u; } v; v.f = f;
  unsigned u = v.u;
  unsigned r = u + 0x7FFFu + ((u >> 16) & 1u);   // round-to-nearest-even
  return (unsigned short)(r >> 16);
}

// Build a v16bf fragment from two 16-byte chunks.
static __device__ __forceinline__ v16bf load_pair(const unsigned short* p0,
                                                  const unsigned short* p1) {
  union { v16bf v; uint4 q[2]; } u;
  u.q[0] = *reinterpret_cast<const uint4*>(p0);
  u.q[1] = *reinterpret_cast<const uint4*>(p1);
  return u.v;
}

// A-fragment (16x32 bf16, M=lane%16): 16 contiguous k at k0 = lane<16?0:16
static __device__ __forceinline__ v16bf load_afrag(const unsigned short* row_ptr, int hi) {
  const int k0 = hi ? 16 : 0;
  return load_pair(row_ptr + k0, row_ptr + k0 + 8);
}

// B-fragment (32x16 bf16, N=lane%16): 16 contiguous k at k0 = lane<16?0:16
static __device__ __forceinline__ v16bf load_bfrag(const unsigned short* row_ptr, int hi) {
  const int k0 = hi ? 16 : 0;
  return load_pair(row_ptr + k0, row_ptr + k0 + 8);
}

// Async global -> LDS copy of 16 bytes per lane (CDNA5 async path, ASYNCcnt).
static __device__ __forceinline__ void async_copy_b128(void* lds_dst, const void* gsrc) {
  unsigned ldsa = (unsigned)(uintptr_t)lds_dst;   // flat LDS aperture: low 32 bits = offset
  asm volatile("global_load_async_to_lds_b128 %0, %1, off"
               :: "v"(ldsa), "v"(gsrc) : "memory");
}

static __device__ __forceinline__ void async_wait0() {
  asm volatile("s_wait_asynccnt 0" ::: "memory");
}
// Leave the newest 4 per-thread async copies (one double-buffered tile) in flight.
static __device__ __forceinline__ void async_wait4() {
  asm volatile("s_wait_asynccnt 4" ::: "memory");
}

// -------------------------------------------------- GEMM: C = A @ W^T + b ---
// A [8192,1024] f32, W [1024,1024] f32 (row n contiguous over k), bias [1024].
// MODE 0: head-major bf16 out [B,H,N,DK].
// MODE 1: transposed bf16 out [B,D,N].
// MODE 2: f32 out [B*N,D].
template <int MODE>
__global__ __launch_bounds__(256) void gemm_xwT_kernel(
    const float* __restrict__ A, const float* __restrict__ W,
    const float* __restrict__ bias, float scale, void* __restrict__ outp) {
  __shared__ __align__(16) unsigned short As[128 * 32];
  __shared__ __align__(16) unsigned short Bs[128 * 32];

  const int t = threadIdx.x;
  const int lane = t & 31, hi = lane >> 4, lo = lane & 15;
  const int w = t >> 5;
  const int mw = (w >> 2) * 64;   // wave M offset: 0 / 64
  const int nw = (w & 3) * 32;    // wave N offset: 0..96
  const int mbase = blockIdx.y * 128;
  const int nbase = blockIdx.x * 128;

  v8f zero = {};
  v8f acc[4][2];
#pragma unroll
  for (int ms = 0; ms < 4; ms++)
#pragma unroll
    for (int ns = 0; ns < 2; ns++) acc[ms][ns] = zero;

  for (int kb = 0; kb < Dd; kb += 32) {
    // cooperative load of A and W tiles (128x32 each), f32 -> bf16 into LDS
#pragma unroll
    for (int i = 0; i < 4; i++) {
      int idx = t + i * 256;            // float4 chunk id in [0,1024)
      int row = idx >> 3;
      int kq  = (idx & 7) << 2;
      float4 fa = *reinterpret_cast<const float4*>(A + (size_t)(mbase + row) * Dd + kb + kq);
      float4 fw = *reinterpret_cast<const float4*>(W + (size_t)(nbase + row) * Dd + kb + kq);
      uint2 pa, pw;
      pa.x = (unsigned)f2bf(fa.x) | ((unsigned)f2bf(fa.y) << 16);
      pa.y = (unsigned)f2bf(fa.z) | ((unsigned)f2bf(fa.w) << 16);
      pw.x = (unsigned)f2bf(fw.x) | ((unsigned)f2bf(fw.y) << 16);
      pw.y = (unsigned)f2bf(fw.z) | ((unsigned)f2bf(fw.w) << 16);
      *reinterpret_cast<uint2*>(&As[row * 32 + kq]) = pa;
      *reinterpret_cast<uint2*>(&Bs[row * 32 + kq]) = pw;
    }
    __syncthreads();

    v16bf af[4], bf[2];
#pragma unroll
    for (int ms = 0; ms < 4; ms++)
      af[ms] = load_afrag(&As[(mw + ms * 16 + lo) * 32], hi);
#pragma unroll
    for (int ns = 0; ns < 2; ns++)
      bf[ns] = load_bfrag(&Bs[(nw + ns * 16 + lo) * 32], hi);
#pragma unroll
    for (int ms = 0; ms < 4; ms++)
#pragma unroll
      for (int ns = 0; ns < 2; ns++)
        acc[ms][ns] = WMMA_BF16(af[ms], bf[ns], acc[ms][ns]);
    __syncthreads();
  }

  // epilogue: C fragment row = i + 8*hi, col = lane%16
#pragma unroll
  for (int ms = 0; ms < 4; ms++) {
#pragma unroll
    for (int ns = 0; ns < 2; ns++) {
      int colg = nbase + nw + ns * 16 + lo;
      float bv = bias[colg];
#pragma unroll
      for (int i = 0; i < 8; i++) {
        int rowg = mbase + mw + ms * 16 + i + 8 * hi;
        float v = (acc[ms][ns][i] + bv) * scale;
        int bb = rowg >> 9, tok = rowg & 511;
        if (MODE == 0) {   // head-major [B,H,N,DK]
          int hh = colg >> 6, dk = colg & 63;
          ((unsigned short*)outp)[((size_t)((bb * Hh + hh) * Nn + tok)) * DKk + dk] = f2bf(v);
        } else if (MODE == 1) {  // transposed [B,D,N]
          ((unsigned short*)outp)[((size_t)bb * Dd + colg) * Nn + tok] = f2bf(v);
        } else {                 // f32 [B*N,D]
          ((float*)outp)[(size_t)rowg * Dd + colg] = v;
        }
      }
    }
  }
}

// ------------------------------------ G = 0.3*softmax(-dist,mask) + 0.4*padj
static __device__ __forceinline__ float breduce_max(float v, float* red) {
  int t = threadIdx.x;
  red[t] = v; __syncthreads();
#pragma unroll
  for (int s = 128; s > 0; s >>= 1) {
    if (t < s) red[t] = fmaxf(red[t], red[t + s]);
    __syncthreads();
  }
  float r = red[0]; __syncthreads();
  return r;
}

static __device__ __forceinline__ float breduce_sum(float v, float* red) {
  int t = threadIdx.x;
  red[t] = v; __syncthreads();
#pragma unroll
  for (int s = 128; s > 0; s >>= 1) {
    if (t < s) red[t] = red[t] + red[t + s];
    __syncthreads();
  }
  float r = red[0]; __syncthreads();
  return r;
}

__global__ __launch_bounds__(256) void build_g_kernel(
    const float* __restrict__ dist, const float* __restrict__ adj,
    const int* __restrict__ mask, unsigned short* __restrict__ G) {
  __shared__ float red[256];
  const int t = threadIdx.x;
  const int r = blockIdx.x;        // row over [0, B*N)
  const int b = r >> 9;
  const float* drow = dist + (size_t)r * Nn;
  const float* arow = adj  + (size_t)r * Nn;
  const int*   mrow = mask + b * Nn;
  const int c0 = t, c1 = t + 256;

  float d0 = drow[c0], d1 = drow[c1];
  float a0 = arow[c0], a1 = arow[c1];
  int   v0 = mrow[c0], v1 = mrow[c1];
  float ninf = -__builtin_inff();
  float n0 = v0 ? -d0 : ninf;
  float n1 = v1 ? -d1 : ninf;
  float mx = breduce_max(fmaxf(n0, n1), red);
  float e0 = v0 ? __expf(n0 - mx) : 0.0f;
  float e1 = v1 ? __expf(n1 - mx) : 0.0f;
  float se = breduce_sum(e0 + e1, red);
  float sa = breduce_sum(a0 + a1, red);
  float ws = se > 0.0f ? 0.3f / se : 0.0f;
  float wa = 0.4f / (sa + 1e-6f);
  G[(size_t)r * Nn + c0] = f2bf(e0 * ws + a0 * wa);
  G[(size_t)r * Nn + c1] = f2bf(e1 * ws + a1 * wa);
}

// -------------------------------------------- X = G @ V (batched, f32 out) --
// A = G[b] (bf16, row-major over k), B = Vt[b] ([D,N]: d rows contiguous over
// tokens) -> both tiles are straight contiguous copies, double-buffered async.
__global__ __launch_bounds__(256) void gv_kernel(
    const unsigned short* __restrict__ G,   // [B,N,N] bf16
    const unsigned short* __restrict__ Vt,  // [B,D,N] bf16
    float* __restrict__ X) {                // [B*N, D] f32
  __shared__ __align__(16) unsigned short As[2][128 * 32];
  __shared__ __align__(16) unsigned short Bs[2][128 * 32];  // [d_local][k_local]

  const int t = threadIdx.x;
  const int lane = t & 31, hi = lane >> 4, lo = lane & 15;
  const int w = t >> 5;
  const int mw = (w >> 2) * 64;
  const int nw = (w & 3) * 32;
  const int nbase = blockIdx.x * 128;   // d
  const int mbase = blockIdx.y * 128;   // q
  const int b = blockIdx.z;
  const unsigned short* Gb = G + (size_t)b * Nn * Nn;
  const unsigned short* Vb = Vt + (size_t)b * Dd * Nn;

  // 4 async issues per thread per tile
  auto issue = [&](int kb, int buf) {
#pragma unroll
    for (int i = 0; i < 2; i++) {
      int idx = t + i * 256;            // uint4 (8-short) chunk id in [0,512)
      int row = idx >> 2, kc = (idx & 3) << 3;
      async_copy_b128(&As[buf][row * 32 + kc],
                      Gb + (size_t)(mbase + row) * Nn + kb + kc);
      async_copy_b128(&Bs[buf][row * 32 + kc],
                      Vb + (size_t)(nbase + row) * Nn + kb + kc);
    }
  };

  v8f zero = {};
  v8f acc[4][2];
#pragma unroll
  for (int ms = 0; ms < 4; ms++)
#pragma unroll
    for (int ns = 0; ns < 2; ns++) acc[ms][ns] = zero;

  issue(0, 0);
  for (int it = 0; it < Nn / 32; it++) {
    const int cur = it & 1;
    if (it + 1 < Nn / 32) { issue((it + 1) * 32, 1 - cur); async_wait4(); }
    else                  { async_wait0(); }
    __syncthreads();      // tile `cur` visible to all waves

    v16bf af[4], bf[2];
#pragma unroll
    for (int ms = 0; ms < 4; ms++)
      af[ms] = load_afrag(&As[cur][(mw + ms * 16 + lo) * 32], hi);
#pragma unroll
    for (int ns = 0; ns < 2; ns++)
      bf[ns] = load_bfrag(&Bs[cur][(nw + ns * 16 + lo) * 32], hi);
#pragma unroll
    for (int ms = 0; ms < 4; ms++)
#pragma unroll
      for (int ns = 0; ns < 2; ns++)
        acc[ms][ns] = WMMA_BF16(af[ms], bf[ns], acc[ms][ns]);
    __syncthreads();      // all reads of tile `cur` done before it is refilled
  }

#pragma unroll
  for (int ms = 0; ms < 4; ms++)
#pragma unroll
    for (int ns = 0; ns < 2; ns++) {
      int colg = nbase + nw + ns * 16 + lo;
#pragma unroll
      for (int i = 0; i < 8; i++) {
        int rowg = mbase + mw + ms * 16 + i + 8 * hi;
        X[((size_t)b * Nn + rowg) * Dd + colg] = acc[ms][ns][i];
      }
    }
}

// --------------------------- flash attention: X += 0.3 * softmax(QK^T) @ V --
// Q prescaled by 1/sqrt(DK). Q/K head-major [B,H,N,DK]; V transposed [B,D,N].
__global__ __launch_bounds__(256) void attn_kernel(
    const unsigned short* __restrict__ Qbf,
    const unsigned short* __restrict__ Kbf,
    const unsigned short* __restrict__ Vt,
    const int* __restrict__ mask,
    float* __restrict__ X) {
  __shared__ __align__(16) unsigned short Ks[2][64 * 64];   // [tok_local][dk]
  __shared__ __align__(16) unsigned short Vs[2][64 * 64];   // [dk][tok_local]
  __shared__ __align__(16) unsigned short Ps[8 * 16 * 32];
  __shared__ int Ms[Nn];

  const int t = threadIdx.x;
  const int lane = t & 31, hi = lane >> 4, lo = lane & 15;
  const int w = t >> 5;
  const int qtile = blockIdx.x & 3;
  const int bh = blockIdx.x >> 2;
  const int h = bh & 15, b = bh >> 4;

  const unsigned short* Kh = Kbf + (size_t)bh * Nn * DKk;
  const unsigned short* Vh = Vt + ((size_t)b * Dd + h * DKk) * Nn;  // [64 dk][512 tok]

  Ms[t] = mask[b * Nn + t];
  Ms[t + 256] = mask[b * Nn + t + 256];

  // 4 async issues per thread per tile (K rows + V rows, both contiguous)
  auto issue = [&](int kt, int buf) {
#pragma unroll
    for (int i = 0; i < 2; i++) {
      int idx = t + i * 256;             // [0,512) uint4 chunks
      int tok = idx >> 3, d8 = (idx & 7) << 3;     // K tile geometry
      async_copy_b128(&Ks[buf][tok * 64 + d8],
                      Kh + (size_t)(kt + tok) * DKk + d8);
      int dk = idx >> 3, t8 = (idx & 7) << 3;      // V tile geometry
      async_copy_b128(&Vs[buf][dk * 64 + t8],
                      Vh + (size_t)dk * Nn + kt + t8);
    }
  };

  // Q strip A-fragments (16 rows x 64 dk) held in registers
  const int qrow = qtile * 128 + w * 16 + lo;
  const unsigned short* Qrow = Qbf + ((size_t)bh * Nn + qrow) * DKk;
  v16bf qa0 = load_afrag(Qrow, hi);        // dk 0..31
  v16bf qa1 = load_afrag(Qrow + 32, hi);   // dk 32..63

  v8f zero = {};
  v8f o0 = zero, o1 = zero, o2 = zero, o3 = zero;
  float m_i[8], l_i[8];
#pragma unroll
  for (int i = 0; i < 8; i++) { m_i[i] = -__builtin_inff(); l_i[i] = 0.0f; }

  unsigned short* Pw = &Ps[w * 512];

  issue(0, 0);
  for (int it = 0; it < Nn / 64; it++) {
    const int cur = it & 1;
    const int kt = it * 64;
    if (it + 1 < Nn / 64) { issue(kt + 64, 1 - cur); async_wait4(); }
    else                  { async_wait0(); }
    __syncthreads();

#pragma unroll
    for (int sub = 0; sub < 2; sub++) {
      const int kb = sub * 32;         // local key base
      const int kg = kt + kb;          // global key base

      // scores S[16q x 32k] = Q @ K^T  (two 16-col fragments)
      v8f s0 = zero, s1 = zero;
      {
        const unsigned short* r0 = &Ks[cur][(kb + lo) * 64];
        s0 = WMMA_BF16(qa0, load_bfrag(r0, hi), s0);
        s0 = WMMA_BF16(qa1, load_bfrag(r0 + 32, hi), s0);
        const unsigned short* r1 = &Ks[cur][(kb + 16 + lo) * 64];
        s1 = WMMA_BF16(qa0, load_bfrag(r1, hi), s1);
        s1 = WMMA_BF16(qa1, load_bfrag(r1 + 32, hi), s1);
      }
      const int mv0 = Ms[kg + lo];
      const int mv1 = Ms[kg + 16 + lo];

      float sc[8];
#pragma unroll
      for (int i = 0; i < 8; i++) {
        float v0 = mv0 ? s0[i] : -1e9f;
        float v1 = mv1 ? s1[i] : -1e9f;
        float mb = fmaxf(v0, v1);
        mb = fmaxf(mb, __shfl_xor(mb, 1, 32));
        mb = fmaxf(mb, __shfl_xor(mb, 2, 32));
        mb = fmaxf(mb, __shfl_xor(mb, 4, 32));
        mb = fmaxf(mb, __shfl_xor(mb, 8, 32));
        float mnew = fmaxf(m_i[i], mb);
        sc[i] = __expf(m_i[i] - mnew);
        float p0 = __expf(v0 - mnew);
        float p1 = __expf(v1 - mnew);
        float rs = p0 + p1;
        rs += __shfl_xor(rs, 1, 32);
        rs += __shfl_xor(rs, 2, 32);
        rs += __shfl_xor(rs, 4, 32);
        rs += __shfl_xor(rs, 8, 32);
        l_i[i] = l_i[i] * sc[i] + rs;
        m_i[i] = mnew;
        // stash P (bf16) into per-wave LDS patch for A-fragment reload
        Pw[(i + 8 * hi) * 32 + lo]      = f2bf(p0);
        Pw[(i + 8 * hi) * 32 + 16 + lo] = f2bf(p1);
      }
#pragma unroll
      for (int i = 0; i < 8; i++) {
        o0[i] *= sc[i]; o1[i] *= sc[i]; o2[i] *= sc[i]; o3[i] *= sc[i];
      }

      v16bf pa = load_afrag(&Pw[lo * 32], hi);
      const int tok0 = kb + (hi ? 16 : 0);
      o0 = WMMA_BF16(pa, load_pair(&Vs[cur][(0 * 16 + lo) * 64 + tok0],
                                   &Vs[cur][(0 * 16 + lo) * 64 + tok0] + 8), o0);
      o1 = WMMA_BF16(pa, load_pair(&Vs[cur][(1 * 16 + lo) * 64 + tok0],
                                   &Vs[cur][(1 * 16 + lo) * 64 + tok0] + 8), o1);
      o2 = WMMA_BF16(pa, load_pair(&Vs[cur][(2 * 16 + lo) * 64 + tok0],
                                   &Vs[cur][(2 * 16 + lo) * 64 + tok0] + 8), o2);
      o3 = WMMA_BF16(pa, load_pair(&Vs[cur][(3 * 16 + lo) * 64 + tok0],
                                   &Vs[cur][(3 * 16 + lo) * 64 + tok0] + 8), o3);
    }
    __syncthreads();   // all reads of tile `cur` done before it is refilled
  }

  // X += 0.3 * O / l   (X already holds the G@V term; ordering via stream)
#pragma unroll
  for (int i = 0; i < 8; i++) {
    float inv = 0.3f / l_i[i];
    int rowg = qtile * 128 + w * 16 + i + 8 * hi;
    float* xp = X + ((size_t)b * Nn + rowg) * Dd + h * DKk + lo;
    xp[0]  += o0[i] * inv;
    xp[16] += o1[i] * inv;
    xp[32] += o2[i] * inv;
    xp[48] += o3[i] * inv;
  }
}

// ------------------------------------------------------------------ launch --

extern "C" void kernel_launch(void* const* d_in, const int* in_sizes, int n_in,
                              void* d_out, int out_size, void* d_ws, size_t ws_size,
                              hipStream_t stream) {
  (void)in_sizes; (void)n_in; (void)out_size; (void)ws_size;

  const float* query = (const float*)d_in[0];
  const float* key   = (const float*)d_in[1];
  const float* value = (const float*)d_in[2];
  const float* adj   = (const float*)d_in[3];
  const float* dist  = (const float*)d_in[4];
  const int*   mask  = (const int*)d_in[5];
  const float* Wq = (const float*)d_in[6];  const float* bq = (const float*)d_in[7];
  const float* Wk = (const float*)d_in[8];  const float* bk = (const float*)d_in[9];
  const float* Wv = (const float*)d_in[10]; const float* bv = (const float*)d_in[11];
  const float* Wo = (const float*)d_in[12]; const float* bo = (const float*)d_in[13];
  float* out = (float*)d_out;

  char* ws = (char*)d_ws;
  size_t off = 0;
  auto alloc = [&](size_t bytes) {
    char* p = ws + off;
    off += (bytes + 255) & ~(size_t)255;
    return p;
  };
  const size_t elemsQKV = (size_t)Bb * Hh * Nn * DKk;        // 8,388,608
  unsigned short* Qbf = (unsigned short*)alloc(elemsQKV * 2);
  unsigned short* Kbf = (unsigned short*)alloc(elemsQKV * 2);
  unsigned short* Vt  = (unsigned short*)alloc(elemsQKV * 2);   // [B,D,N]
  unsigned short* G   = (unsigned short*)alloc((size_t)Bb * Nn * Nn * 2);
  float*          X   = (float*)alloc((size_t)Bb * Nn * Dd * 4);

  dim3 blk(256);
  dim3 gGemm(Dd / 128, (Bb * Nn) / 128);   // (8, 64)

  // Projections (Q carries the 1/sqrt(DK) scale folded into its bf16 store)
  gemm_xwT_kernel<0><<<gGemm, blk, 0, stream>>>(query, Wq, bq, 0.125f, Qbf);
  gemm_xwT_kernel<0><<<gGemm, blk, 0, stream>>>(key,   Wk, bk, 1.0f,   Kbf);
  gemm_xwT_kernel<1><<<gGemm, blk, 0, stream>>>(value, Wv, bv, 1.0f,   Vt);

  // G = 0.3*softmax(-dist masked) + 0.4*adj/rowsum
  build_g_kernel<<<dim3(Bb * Nn), blk, 0, stream>>>(dist, adj, mask, G);

  // X = G @ V (head-independent part)
  gv_kernel<<<dim3(Dd / 128, Nn / 128, Bb), blk, 0, stream>>>(G, Vt, X);

  // X += 0.3 * softmax(Q K^T) @ V per head
  attn_kernel<<<dim3(Bb * Hh * (Nn / 128)), blk, 0, stream>>>(Qbf, Kbf, Vt, mask, X);

  // out = X @ Wo^T + bo
  gemm_xwT_kernel<2><<<gGemm, blk, 0, stream>>>(X, Wo, bo, 1.0f, out);
}
